// NeuralCDE_28664611733739
// MI455X (gfx1250) — matrix operand order repaired and checked by reading, not compiled, not run on hardware
//
#include <hip/hip_runtime.h>
#include <hip/hip_bf16.h>
#include <cstdint>

// ---------------------------------------------------------------------------
// Neural CDE (reversible Heun) on gfx1250 via v_wmma_f32_16x16x32_f16.
// B=16384, L=128, C=9, HIDDEN=32, MLP=128.
// ---------------------------------------------------------------------------

typedef __attribute__((ext_vector_type(16))) _Float16 v16h;
typedef __attribute__((ext_vector_type(8)))  float    v8f;

#define BSZ   16384
#define LSZ   128
#define CSZ   9
#define HID   32

// ---- workspace layout (bytes) ----
#define WS_H0      0                       // 16384*32*4 = 2097152
#define WS_INITW   2097152                 // 40960 halfs = 81920
#define WS_INITB   (WS_INITW + 81920)      // 416 floats -> pad 2048
#define WS_FUNCW   (WS_INITB + 2048)       // 77824 halfs = 155648
#define WS_FUNCB   (WS_FUNCW + 155648)     // 672 floats

// packed half offsets inside initW / funcW
#define IW_L1 0
#define IW_L2 4096
#define IW_L3 20480
#define IW_L4 36864
#define FW_L1 0
#define FW_L2 8192
#define FW_L3 24576
#define FW_L4 40960

#define ACT_PITCH 136     // halfs per row in activation buffers (conflict-free)
#define FOUT_PITCH 296    // halfs per row in final-layer tanh buffer

// ---------------------------------------------------------------------------
// fast device math: single-instruction rcp / tanh where available
// ---------------------------------------------------------------------------
__device__ __forceinline__ float fast_rcp(float x) {
#if __has_builtin(__builtin_amdgcn_rcpf)
    return __builtin_amdgcn_rcpf(x);
#else
    return 1.0f / x;
#endif
}
__device__ __forceinline__ float fast_tanh(float x) {
#if __has_builtin(__builtin_amdgcn_tanhf)
    return __builtin_amdgcn_tanhf(x);
#else
    float e = __expf(-2.0f * x);
    return (1.0f - e) * fast_rcp(1.0f + e);
#endif
}
__device__ __forceinline__ float lipswish_f(float x) {
    return 0.909f * x * fast_rcp(1.0f + __expf(-x));
}

// ---------------------------------------------------------------------------
// Async global->LDS copy (CDNA5 GLOBAL_LOAD_ASYNC_TO_LDS_B128, ASYNCcnt).
// LDS operand = low 32 bits of the generic shared-space address (ISA flat
// addressing truncates generic LDS addresses to the 32-bit LDS offset).
// ---------------------------------------------------------------------------
__device__ __forceinline__ void async_copy_b128(const void* gsrc, void* ldst) {
    uint32_t l = (uint32_t)(uintptr_t)ldst;
    uint64_t a = (uint64_t)(uintptr_t)gsrc;
    asm volatile("global_load_async_to_lds_b128 %0, %1, off"
                 :: "v"(l), "v"(a) : "memory");
}
__device__ __forceinline__ void wait_asynccnt0() {
#if __has_builtin(__builtin_amdgcn_s_wait_asynccnt)
    __builtin_amdgcn_s_wait_asynccnt(0);
#else
    asm volatile("s_wait_asynccnt 0" ::: "memory");
#endif
}

// ---------------------------------------------------------------------------
// Weight packing: fp32 W[N][Kreal] -> f16 chunks of 512 halves.
// chunk(nt,kb): element (lane*16+e) = W[nt*16 + lane%16][kb*32 + (e<8 ? 8*(lane/16)+e : 16+8*(lane/16)+e-8)]
// matching the documented wave32 B-operand layout for V_WMMA_*_16x16x32_F16.
// ---------------------------------------------------------------------------
__global__ void pack_layer(const float* __restrict__ W, const float* __restrict__ bsrc,
                           _Float16* __restrict__ dstW, float* __restrict__ dstB,
                           int Kreal, int Kp, int N) {
    int total = Kp * N;
    int t = blockIdx.x * 256 + threadIdx.x;
    if (t < total) {
        int chunk = t >> 9;
        int r     = t & 511;
        int lane  = r >> 4;
        int e     = r & 15;
        int KB    = Kp >> 5;
        int nt    = chunk / KB;
        int kb    = chunk - nt * KB;
        int n = nt * 16 + (lane & 15);
        int k = kb * 32 + ((e < 8) ? (((lane >> 4) << 3) + e) : (16 + ((lane >> 4) << 3) + (e - 8)));
        float v = (k < Kreal) ? W[n * Kreal + k] : 0.0f;
        dstW[t] = (_Float16)v;
    }
    if (t < N) dstB[t] = bsrc[t];
}

// ---------------------------------------------------------------------------
// A-operand load: 16-bit A 16x32 layout.
// ---------------------------------------------------------------------------
__device__ __forceinline__ v16h load_A(const _Float16* buf, int lane, int kb) {
    const int m = lane & 15;
    const int g = (lane >> 4) & 1;
    v16h a;
    int* ai = reinterpret_cast<int*>(&a);
    const char* base = reinterpret_cast<const char*>(buf)
                     + m * (ACT_PITCH * 2) + (kb * 32 + 8 * g) * 2;
#pragma unroll
    for (int v = 0; v < 8; ++v) {
        int kpair = (v < 4) ? (2 * v) : (16 + 2 * (v - 4));
        ai[v] = *reinterpret_cast<const int*>(base + kpair * 2);
    }
    return a;
}

// ---------------------------------------------------------------------------
// One MLP layer on a 16-row group: out = act(in[16][32*KB] @ W^T + b).
// A operand cached in registers once; tiles [t0,t1) of 16 output columns.
// ACT: 0 = lipswish, 1 = tanh.
// ---------------------------------------------------------------------------
template<int ACT, int KB>
__device__ __forceinline__ void mlp_layer(const _Float16* inBuf, _Float16* outBuf,
                                          int outPitch, const _Float16* Wl,
                                          const float* bias,
                                          int t0, int t1, int lane) {
    const int m = lane & 15;
    const int g = (lane >> 4) & 1;
    v16h aC[KB];
#pragma unroll
    for (int kb = 0; kb < KB; ++kb) aC[kb] = load_A(inBuf, lane, kb);

    for (int nt = t0; nt < t1; ++nt) {
        float bv = bias[nt * 16 + m];
        v8f c;
#pragma unroll
        for (int r = 0; r < 8; ++r) c[r] = bv;
#pragma unroll
        for (int kb = 0; kb < KB; ++kb) {
            v16h b = *reinterpret_cast<const v16h*>(Wl + (nt * KB + kb) * 512 + lane * 16);
            c = __builtin_amdgcn_wmma_f32_16x16x32_f16(false, aC[kb], false, b,
                                                       (short)0, c, false, false);
        }
#pragma unroll
        for (int r = 0; r < 8; ++r) {
            float x = c[r];
            float o = (ACT == 0) ? lipswish_f(x) : fast_tanh(x);
            outBuf[(r + 8 * g) * outPitch + nt * 16 + m] = (_Float16)o;
        }
    }
}

// ---------------------------------------------------------------------------
// Init MLP: h0 = MLP_init(ys[:,0,:]).  8 waves/block, one 16-row group each.
// LDS: 81920 (weights) + 2048 (bias) + 8 * 8704 (bufX/bufY) = 153600 bytes.
// ---------------------------------------------------------------------------
extern "C" __global__ void __launch_bounds__(256, 1)
cde_init(const float* __restrict__ ys, const _Float16* __restrict__ initW_g,
         const float* __restrict__ initB_g, float* __restrict__ h0) {
    extern __shared__ char smem[];
    _Float16* Wl = reinterpret_cast<_Float16*>(smem);
    float*    Bi = reinterpret_cast<float*>(smem + 81920);
    char*     gb = smem + 81920 + 2048;

    const int tid  = threadIdx.x;
    const int wave = tid >> 5;
    const int lane = tid & 31;
    const int rowBase = blockIdx.x * 128 + wave * 16;

    {   // async weight preload (global -> LDS, no VGPR staging)
        const uint4* src = reinterpret_cast<const uint4*>(initW_g);
        uint4* dst = reinterpret_cast<uint4*>(Wl);
        for (int i = tid; i < 81920 / 16; i += 256) async_copy_b128(src + i, dst + i);
        for (int i = tid; i < 416; i += 256) Bi[i] = initB_g[i];
        wait_asynccnt0();
    }
    __syncthreads();

    _Float16* BX = reinterpret_cast<_Float16*>(gb + wave * 8704);
    _Float16* BY = BX + 2176;   // 16*136

    // build input: cols 0..8 = ys[:,0,:], cols 9..31 = 0  (K padded to 32)
    for (int i = lane; i < 16 * 32; i += 32) {
        int row = i >> 5, cc = i & 31;
        float v = (cc < CSZ) ? ys[(size_t)(rowBase + row) * LSZ * CSZ + cc] : 0.0f;
        BX[row * ACT_PITCH + cc] = (_Float16)v;
    }

    mlp_layer<0, 1>(BX, BY, ACT_PITCH, Wl + IW_L1, Bi + 0,   0, 8, lane);
    mlp_layer<0, 4>(BY, BX, ACT_PITCH, Wl + IW_L2, Bi + 128, 0, 8, lane);
    mlp_layer<0, 4>(BX, BY, ACT_PITCH, Wl + IW_L3, Bi + 256, 0, 8, lane);

    // final linear 128->32, no activation, write f32 directly to h0
    {
        const int m = lane & 15, g = (lane >> 4) & 1;
        v16h aC[4];
#pragma unroll
        for (int kb = 0; kb < 4; ++kb) aC[kb] = load_A(BY, lane, kb);
        for (int nt = 0; nt < 2; ++nt) {
            float bv = Bi[384 + nt * 16 + m];
            v8f c;
#pragma unroll
            for (int r = 0; r < 8; ++r) c[r] = bv;
#pragma unroll
            for (int kb = 0; kb < 4; ++kb) {
                v16h b = *reinterpret_cast<const v16h*>(Wl + IW_L4 + (nt * 4 + kb) * 512 + lane * 16);
                c = __builtin_amdgcn_wmma_f32_16x16x32_f16(false, aC[kb], false, b,
                                                           (short)0, c, false, false);
            }
#pragma unroll
            for (int r = 0; r < 8; ++r)
                h0[(size_t)(rowBase + r + 8 * g) * HID + nt * 16 + m] = c[r];
        }
    }
}

// ---------------------------------------------------------------------------
// Main time loop. 256 threads = 8 waves; 4 groups of 16 rows, 2 waves/group.
// LDS map (bytes):
//   0        func weights f16                 155648
//   155648   func biases  f32 (672, pad 2816)
//   158464   4 groups * 25856:
//            BX 4352 | BY 4352 | FOUT 9472 (L2 scratch aliases its start)
//            | Y 2048 | YH 2048 | FH 2048 | SL 2*768
// total 261888
// ---------------------------------------------------------------------------
#define G_STRIDE 25856
#define O_BX   0
#define O_BY   4352
#define O_FOUT 8704
#define O_Y    18176
#define O_YH   20224
#define O_FH   22272
#define O_SL   24320

extern "C" __global__ void __launch_bounds__(256, 1)
cde_main(const float* __restrict__ ys, const _Float16* __restrict__ funcW_g,
         const float* __restrict__ funcB_g, const float* __restrict__ h0,
         const float* __restrict__ readW, const float* __restrict__ readB,
         float* __restrict__ out) {
    extern __shared__ char smem[];
    _Float16* Wl = reinterpret_cast<_Float16*>(smem);
    float*    Bi = reinterpret_cast<float*>(smem + 155648);
    char*     groupsBase = smem + 158464;

    const int tid  = threadIdx.x;
    const int wave = tid >> 5;
    const int lane = tid & 31;
    const int g    = wave >> 1;     // group 0..3
    const int half = wave & 1;      // N-tile split within group
    const int tig  = tid & 63;      // thread index within group (64 threads)
    const int rowBase = blockIdx.x * 64 + g * 16;

    {   // async weight preload (152KB global -> LDS, once per block)
        const uint4* src = reinterpret_cast<const uint4*>(funcW_g);
        uint4* dst = reinterpret_cast<uint4*>(Wl);
        for (int i = tid; i < 155648 / 16; i += 256) async_copy_b128(src + i, dst + i);
        for (int i = tid; i < 672; i += 256) Bi[i] = funcB_g[i];
        wait_asynccnt0();
    }

    char* gb = groupsBase + g * G_STRIDE;
    _Float16* BX   = reinterpret_cast<_Float16*>(gb + O_BX);
    _Float16* BY   = reinterpret_cast<_Float16*>(gb + O_BY);
    _Float16* ZZ   = reinterpret_cast<_Float16*>(gb + O_FOUT);   // L2 out (pitch 136)
    _Float16* FOUT = reinterpret_cast<_Float16*>(gb + O_FOUT);   // L4 out (pitch 296)
    float* Y  = reinterpret_cast<float*>(gb + O_Y);
    float* YH = reinterpret_cast<float*>(gb + O_YH);
    float* FH = reinterpret_cast<float*>(gb + O_FH);

    // initial state y = yhat = h0 ; zero th pad cols 33..63 once
    for (int i = tig; i < 16 * HID; i += 64) {
        int row = i >> 5, h = i & 31;
        float v = h0[(size_t)(rowBase + row) * HID + h];
        Y[i] = v; YH[i] = v;
    }
    for (int i = tig; i < 16 * 31; i += 64) {
        int row = i / 31, cc = 33 + (i - row * 31);
        BX[row * ACT_PITCH + cc] = (_Float16)0.0f;
    }
    __syncthreads();

    // s = 0: f0 = drift(0, h0, slope0); s >= 1: full reversible-Heun step.
    for (int s = 0; s < LSZ; ++s) {
        const int seg = (s < LSZ - 2) ? s : (LSZ - 2);
        float* SL = reinterpret_cast<float*>(gb + O_SL + (s & 1) * 768);

        // slopes for this segment (double-buffered -> no barrier vs einsum)
        for (int i = tig; i < 16 * CSZ; i += 64) {
            int row = i / CSZ, c = i - row * CSZ;
            const float* p = ys + (size_t)(rowBase + row) * LSZ * CSZ + seg * CSZ + c;
            SL[row * 12 + c] = p[CSZ] - p[0];
        }
        // prefetch next segment's rows; latency hidden behind this step's WMMAs
        {
            const int segN = (s + 1 < LSZ - 2) ? (s + 1) : (LSZ - 2);
            for (int i = tig; i < 16; i += 64) {
                const char* pp = (const char*)(ys + (size_t)(rowBase + i) * LSZ * CSZ + segN * CSZ);
                __builtin_prefetch(pp, 0, 0);
                __builtin_prefetch(pp + 64, 0, 0);
            }
        }

        // build th = [t, hvec] into BX cols 0..32; update yhat in place
        const float tval = (float)s;
        for (int i = tig; i < 16 * HID; i += 64) {
            int row = i >> 5, h = i & 31;
            float hv;
            if (s == 0) hv = Y[i];
            else { hv = 2.0f * Y[i] - YH[i] + FH[i]; YH[i] = hv; }
            BX[row * ACT_PITCH + 1 + h] = (_Float16)hv;
        }
        if (tig < 16) BX[tig * ACT_PITCH] = (_Float16)tval;
        __syncthreads();

        // drift MLP: 33->128->128->128->288, lipswish x3, tanh
        mlp_layer<0, 2>(BX, BY, ACT_PITCH,  Wl + FW_L1, Bi + 0,   half * 4, half * 4 + 4, lane);
        __syncthreads();
        mlp_layer<0, 4>(BY, ZZ, ACT_PITCH,  Wl + FW_L2, Bi + 128, half * 4, half * 4 + 4, lane);
        __syncthreads();
        mlp_layer<0, 4>(ZZ, BY, ACT_PITCH,  Wl + FW_L3, Bi + 256, half * 4, half * 4 + 4, lane);
        __syncthreads();
        mlp_layer<1, 4>(BY, FOUT, FOUT_PITCH, Wl + FW_L4, Bi + 384, half * 9, half * 9 + 9, lane);
        __syncthreads();

        // einsum + Heun update fused (Y/YH/FH indices are thread-private)
        for (int i = tig; i < 16 * HID; i += 64) {
            int row = i >> 5, h = i & 31;
            float acc = 0.0f;
#pragma unroll
            for (int c = 0; c < CSZ; ++c)
                acc += (float)FOUT[row * FOUT_PITCH + h * CSZ + c] * SL[row * 12 + c];
            if (s == 0) FH[i] = acc;
            else { Y[i] += 0.5f * (FH[i] + acc); FH[i] = acc; }
        }
    }
    __syncthreads();

    // readout row 0: out[b] = y[b] . readW[0,:] + readB[0]
    if (tig < 16) {
        float acc = readB[0];
#pragma unroll
        for (int h = 0; h < HID; ++h) acc += Y[tig * HID + h] * readW[h];
        out[rowBase + tig] = acc;
    }
}

// ---------------------------------------------------------------------------
// Host launcher
// ---------------------------------------------------------------------------
extern "C" void kernel_launch(void* const* d_in, const int* in_sizes, int n_in,
                              void* d_out, int out_size, void* d_ws, size_t ws_size,
                              hipStream_t stream) {
    const float* ys = (const float*)d_in[0];
    const float* readW = (const float*)d_in[17];
    const float* readB = (const float*)d_in[18];

    char* ws = (char*)d_ws;
    float*    H0    = (float*)(ws + WS_H0);
    _Float16* initW = (_Float16*)(ws + WS_INITW);
    float*    initB = (float*)(ws + WS_INITB);
    _Float16* funcW = (_Float16*)(ws + WS_FUNCW);
    float*    funcB = (float*)(ws + WS_FUNCB);

    // ---- pack weights to f16 WMMA-B layout ----
    pack_layer<<<16,  256, 0, stream>>>((const float*)d_in[1], (const float*)d_in[2],
                                        initW + IW_L1, initB + 0,   9,   32, 128);
    pack_layer<<<64,  256, 0, stream>>>((const float*)d_in[3], (const float*)d_in[4],
                                        initW + IW_L2, initB + 128, 128, 128, 128);
    pack_layer<<<64,  256, 0, stream>>>((const float*)d_in[5], (const float*)d_in[6],
                                        initW + IW_L3, initB + 256, 128, 128, 128);
    pack_layer<<<16,  256, 0, stream>>>((const float*)d_in[7], (const float*)d_in[8],
                                        initW + IW_L4, initB + 384, 128, 128, 32);
    pack_layer<<<32,  256, 0, stream>>>((const float*)d_in[9],  (const float*)d_in[10],
                                        funcW + FW_L1, funcB + 0,   33,  64,  128);
    pack_layer<<<64,  256, 0, stream>>>((const float*)d_in[11], (const float*)d_in[12],
                                        funcW + FW_L2, funcB + 128, 128, 128, 128);
    pack_layer<<<64,  256, 0, stream>>>((const float*)d_in[13], (const float*)d_in[14],
                                        funcW + FW_L3, funcB + 256, 128, 128, 128);
    pack_layer<<<144, 256, 0, stream>>>((const float*)d_in[15], (const float*)d_in[16],
                                        funcW + FW_L4, funcB + 384, 128, 128, 288);

    // ---- h0 = init MLP(ys[:,0,:]) ----
    cde_init<<<BSZ / 128, 256, 153600, stream>>>(ys, initW, initB, H0);

    // ---- time loop + readout ----
    cde_main<<<BSZ / 64, 256, 261888, stream>>>(ys, funcW, funcB, H0,
                                                readW, readB, (float*)d_out);
}
